// STGCN_45440753992503
// MI455X (gfx1250) — matrix operand order repaired and checked by reading, not compiled, not run on hardware
//
#include <hip/hip_runtime.h>
#include <hip/hip_bf16.h>
#include <cstdint>
#include <cstddef>

// ---------------------------------------------------------------------------
// STGCN forward for MI455X (gfx1250, wave32). All heavy GEMMs go through
// v_wmma_f32_16x16x32_bf16 (bf16 inputs, f32 accumulate). Per-lane WMMA
// operand layouts (CDNA5 ISA 7.12.2) are fed directly from global memory
// with 16B contiguous loads -- no LDS staging needed given the chosen
// tensor layouts (c-major (bt,C,N) for spatial, n-major (bt,N,C) for
// temporal convs, K-major bf16 weight matrices).
// ---------------------------------------------------------------------------

typedef __bf16 bf16_t;
typedef __attribute__((ext_vector_type(16))) __bf16 v16bf;
typedef __attribute__((ext_vector_type(8)))  __bf16 v8bf;
typedef __attribute__((ext_vector_type(8)))  float  v8f;

#define NND 1024  // number of graph nodes

__device__ __forceinline__ v8bf ld8(const bf16_t* p) {
  return *reinterpret_cast<const v8bf*>(p);
}
__device__ __forceinline__ v8bf zero8() {
  v8bf z;
#pragma unroll
  for (int i = 0; i < 8; ++i) z[i] = (bf16_t)0.0f;
  return z;
}
__device__ __forceinline__ v16bf mk16(v8bf lo, v8bf hi) {
  v16bf r;
#pragma unroll
  for (int i = 0; i < 8; ++i) { r[i] = lo[i]; r[i + 8] = hi[i]; }
  return r;
}
__device__ __forceinline__ float sigmoidf_(float x) {
  return 1.0f / (1.0f + __expf(-x));
}

// -------------------------------- prep kernels -----------------------------

// f32 -> bf16 bulk convert (cheb: 3*1024*1024)
__global__ void k_cvt_bf16(const float* __restrict__ src,
                           bf16_t* __restrict__ dst, int total) {
  int i = blockIdx.x * blockDim.x + threadIdx.x;
  if (i < total) dst[i] = (bf16_t)src[i];
}

// conv weight (2*Td, Ci, 1, 3) f32 -> B-matrix bf16 [co][Kpad], K index =
// kt*Ci+ci, zero padded to Kpad (multiple of 32).
__global__ void k_prep_w(const float* __restrict__ w, bf16_t* __restrict__ wBt,
                         int Ci, int Kpad) {
  int i = blockIdx.x * blockDim.x + threadIdx.x;
  int total = 128 * Kpad;
  if (i >= total) return;
  int co = i / Kpad, j = i - co * Kpad;
  float v = 0.0f;
  if (j < 3 * Ci) {
    int kt = j / Ci, ci = j - kt * Ci;
    v = w[(co * Ci + ci) * 3 + kt];
  }
  wBt[i] = (bf16_t)v;
}

// ----------------------- temporal conv #1 (Cin=1, VALU) --------------------
// x (B,12,1024,1) f32 -> a1 c-major (bt,64,1024) bf16.  0.25 GFLOP: VALU ok.
__global__ void k_tconv1(const float* __restrict__ x, const float* __restrict__ w,
                         const float* __restrict__ bias, bf16_t* __restrict__ outc,
                         int T_in, int T_out) {
  int n  = blockIdx.x * blockDim.x + threadIdx.x;
  int bt = blockIdx.y;
  int b = bt / T_out, tp = bt - b * T_out;
  size_t xb = (size_t)(b * T_in + tp) * NND + n;
  float x0 = x[xb], x1 = x[xb + NND], x2 = x[xb + 2 * NND];
#pragma unroll 4
  for (int c = 0; c < 64; ++c) {
    float lhs = w[c * 3 + 0] * x0 + w[c * 3 + 1] * x1 + w[c * 3 + 2] * x2 + bias[c];
    int cr = c + 64;
    float rhs = w[cr * 3 + 0] * x0 + w[cr * 3 + 1] * x1 + w[cr * 3 + 2] * x2 + bias[cr];
    outc[((size_t)bt * 64 + c) * NND + n] = (bf16_t)(lhs * sigmoidf_(rhs));
  }
}

// ------------------- temporal gated conv via WMMA (Ci=16/64) ---------------
// in: n-major (bt_in = b*T_in+t, 1024, Ci) bf16; wBt: [128][Kpad] bf16;
// out: gated (bt_out, ...) bf16, n-major and/or c-major.
// Wave: 16 n-rows x (co tile jt and co tile jt+64) -> 2 WMMA accumulators.
__global__ void k_tconv_wmma(const bf16_t* __restrict__ in,
                             const bf16_t* __restrict__ wBt,
                             const float* __restrict__ bias,
                             bf16_t* __restrict__ outn,  // may be null
                             bf16_t* __restrict__ outc,  // may be null
                             int T_in, int T_out, int Ci, int ciShift, int Kpad) {
  int bt = blockIdx.y;
  int b = bt / T_out, tp = bt - b * T_out;
  int bT = b * T_in + tp;  // first input time slice

  int lane = threadIdx.x & 31;
  int wv   = threadIdx.x >> 5;
  int jt   = wv & 3;                         // lhs channel tile 0..3
  int nbase = blockIdx.x * 32 + (wv >> 2) * 16;
  int l16  = lane & 15;
  int hi   = lane >> 4;                      // 0 or 1
  int c0   = hi ? 8 : 0;                     // A chunk base within 32-K step
  int n    = nbase + l16;                    // A row (M index)
  int threeCi = 3 * Ci;
  int colL = jt * 16 + l16;                  // B column (lhs channel)

  v8f accL, accR;
#pragma unroll
  for (int i = 0; i < 8; ++i) { accL[i] = 0.0f; accR[i] = 0.0f; }

  int nk = Kpad >> 5;
  for (int ks = 0; ks < nk; ++ks) {
    // ---- A: row n, K chunks [j0..j0+7] and [j0+16..j0+23] (16B each)
    v8bf alo, ahi;
    {
      int j0 = ks * 32 + c0;
      int kt = j0 >> ciShift, ci = j0 & (Ci - 1);
      alo = (j0 < threeCi)
              ? ld8(in + ((size_t)(bT + kt) * NND + n) * Ci + ci) : zero8();
      int j1 = j0 + 16;
      int kt1 = j1 >> ciShift, ci1 = j1 & (Ci - 1);
      ahi = (j1 < threeCi)
              ? ld8(in + ((size_t)(bT + kt1) * NND + n) * Ci + ci1) : zero8();
    }
    v16bf a = mk16(alo, ahi);
    // ---- B: column co, K chunks [kb..kb+7] and [kb+16..kb+23] (zero padded in prep)
    int kb = ks * 32 + hi * 8;
    const bf16_t* bL = wBt + (size_t)colL * Kpad + kb;
    const bf16_t* bR = wBt + (size_t)(colL + 64) * Kpad + kb;
    v16bf vbL = mk16(ld8(bL), ld8(bL + 16));
    v16bf vbR = mk16(ld8(bR), ld8(bR + 16));
    accL = __builtin_amdgcn_wmma_f32_16x16x32_bf16(false, a, false, vbL,
                                                   (short)0, accL, false, false);
    accR = __builtin_amdgcn_wmma_f32_16x16x32_bf16(false, a, false, vbR,
                                                   (short)0, accR, false, false);
  }

  float bl = bias[colL], br = bias[colL + 64];
#pragma unroll
  for (int r = 0; r < 8; ++r) {
    int m = r + hi * 8;                  // D row (n within tile)
    int nrow = nbase + m;
    float o = (accL[r] + bl) * sigmoidf_(accR[r] + br);
    bf16_t ob = (bf16_t)o;
    if (outn) outn[((size_t)bt * NND + nrow) * 64 + colL] = ob;
    if (outc) outc[((size_t)bt * 64 + colL) * NND + nrow] = ob;
  }
}

// --------------------- spatial GEMM1: pong = cheb[k] @ ping ----------------
// Computed as pong_t(C=64 x N) = ping_t(64 x K=1024) @ cheb_k_rows(K x N).
// ping/pong c-major (bt,64,1024) bf16; cheb_k row-major (1024,1024) bf16.
// Register-blocked: each wave owns a 32(C) x 64(n_out) macro-tile =
// 8 WMMA accumulators; per 32-wide K step: 4 A-loads + 8 B-loads feed
// 8 WMMAs (1.5 loads/WMMA). WG (8 waves) covers 64 C x 256 n_out.
__global__ void k_spmm_wmma(const bf16_t* __restrict__ ping,
                            const bf16_t* __restrict__ chebk,
                            bf16_t* __restrict__ pong) {
  int bt = blockIdx.y;
  int lane = threadIdx.x & 31;
  int wv   = threadIdx.x >> 5;
  int ch2  = (wv & 1) * 32;      // base of this wave's two C tiles
  int q    = wv >> 1;            // n_out quarter 0..3
  int l16  = lane & 15;
  int hi   = lane >> 4;
  int c0   = hi ? 8 : 0;

  const bf16_t* arow0 = ping + ((size_t)bt * 64 + ch2 + l16) * NND;        // c-tile 0
  const bf16_t* arow1 = arow0 + (size_t)16 * NND;                          // c-tile 1
  int n0 = blockIdx.x * 256 + q * 64;
  const bf16_t* brow0 = chebk + (size_t)(n0 + 0 * 16 + l16) * NND + hi * 8;
  const bf16_t* brow1 = chebk + (size_t)(n0 + 1 * 16 + l16) * NND + hi * 8;
  const bf16_t* brow2 = chebk + (size_t)(n0 + 2 * 16 + l16) * NND + hi * 8;
  const bf16_t* brow3 = chebk + (size_t)(n0 + 3 * 16 + l16) * NND + hi * 8;

  v8f a0t0, a0t1, a0t2, a0t3, a1t0, a1t1, a1t2, a1t3;
#pragma unroll
  for (int i = 0; i < 8; ++i) {
    a0t0[i] = 0.f; a0t1[i] = 0.f; a0t2[i] = 0.f; a0t3[i] = 0.f;
    a1t0[i] = 0.f; a1t1[i] = 0.f; a1t2[i] = 0.f; a1t3[i] = 0.f;
  }

  for (int ks = 0; ks < 32; ++ks) {
    int kb = ks * 32;
    // prefetch next K-chunk of the streamed cheb rows
    __builtin_prefetch((const void*)(brow0 + kb + 32), 0, 0);
    __builtin_prefetch((const void*)(brow2 + kb + 32), 0, 0);
    v16bf a0 = mk16(ld8(arow0 + kb + c0), ld8(arow0 + kb + c0 + 16));
    v16bf a1 = mk16(ld8(arow1 + kb + c0), ld8(arow1 + kb + c0 + 16));
    v16bf b0 = mk16(ld8(brow0 + kb), ld8(brow0 + kb + 16));
    v16bf b1 = mk16(ld8(brow1 + kb), ld8(brow1 + kb + 16));
    v16bf b2 = mk16(ld8(brow2 + kb), ld8(brow2 + kb + 16));
    v16bf b3 = mk16(ld8(brow3 + kb), ld8(brow3 + kb + 16));
    a0t0 = __builtin_amdgcn_wmma_f32_16x16x32_bf16(false, a0, false, b0, (short)0, a0t0, false, false);
    a1t0 = __builtin_amdgcn_wmma_f32_16x16x32_bf16(false, a1, false, b0, (short)0, a1t0, false, false);
    a0t1 = __builtin_amdgcn_wmma_f32_16x16x32_bf16(false, a0, false, b1, (short)0, a0t1, false, false);
    a1t1 = __builtin_amdgcn_wmma_f32_16x16x32_bf16(false, a1, false, b1, (short)0, a1t1, false, false);
    a0t2 = __builtin_amdgcn_wmma_f32_16x16x32_bf16(false, a0, false, b2, (short)0, a0t2, false, false);
    a1t2 = __builtin_amdgcn_wmma_f32_16x16x32_bf16(false, a1, false, b2, (short)0, a1t2, false, false);
    a0t3 = __builtin_amdgcn_wmma_f32_16x16x32_bf16(false, a0, false, b3, (short)0, a0t3, false, false);
    a1t3 = __builtin_amdgcn_wmma_f32_16x16x32_bf16(false, a1, false, b3, (short)0, a1t3, false, false);
  }

  int mhi = hi * 8;
#pragma unroll
  for (int r = 0; r < 8; ++r) {
    int ca = ch2 + r + mhi;                     // D row (channel), c-tile 0
    int cb = ca + 16;                           // c-tile 1
    size_t rba = ((size_t)bt * 64 + ca) * NND + n0 + l16;
    size_t rbb = ((size_t)bt * 64 + cb) * NND + n0 + l16;
    pong[rba +  0] = (bf16_t)a0t0[r];
    pong[rba + 16] = (bf16_t)a0t1[r];
    pong[rba + 32] = (bf16_t)a0t2[r];
    pong[rba + 48] = (bf16_t)a0t3[r];
    pong[rbb +  0] = (bf16_t)a1t0[r];
    pong[rbb + 16] = (bf16_t)a1t1[r];
    pong[rbb + 32] = (bf16_t)a1t2[r];
    pong[rbb + 48] = (bf16_t)a1t3[r];
  }
}

// ---------------- spatial GEMM2: outacc += pong @ theta[k] (VALU) ----------
// pong c-major (bt,64,1024) bf16; theta_k (64,16) f32; outacc n-major
// (bt,1024,16) f32. kflag: 0=first(write), 1=add, 2=last(add+relu->bf16).
__global__ void k_theta(const bf16_t* __restrict__ pong,
                        const float* __restrict__ theta_k,
                        float* __restrict__ outacc, bf16_t* __restrict__ sout,
                        int kflag) {
  int s = threadIdx.x & 15;
  int n = blockIdx.x * 16 + (threadIdx.x >> 4);
  int bt = blockIdx.y;
  size_t oidx = ((size_t)bt * NND + n) * 16 + s;
  float acc = (kflag == 0) ? 0.0f : outacc[oidx];
#pragma unroll 8
  for (int c = 0; c < 64; ++c)
    acc += (float)pong[((size_t)bt * 64 + c) * NND + n] * theta_k[c * 16 + s];
  if (kflag == 2) {
    sout[oidx] = (bf16_t)fmaxf(acc, 0.0f);
  } else {
    outacc[oidx] = acc;
  }
}

// ------------------------------- final FC ----------------------------------
// af n-major (b*2+t, 1024, 64) bf16; x_emb (1024,96) f32; fc_w (12,224) f32.
// out f32 (B,12,1024,1).
__global__ void k_fc(const bf16_t* __restrict__ af, const float* __restrict__ emb,
                     const float* __restrict__ fcw, const float* __restrict__ fcb,
                     float* __restrict__ out) {
  int idx = blockIdx.x * blockDim.x + threadIdx.x;
  int n = idx & (NND - 1);
  int rem = idx >> 10;
  int p = rem % 12;
  int b = rem / 12;
  const float* wr = fcw + p * 224;
  float acc = fcb[p];
#pragma unroll 2
  for (int t = 0; t < 2; ++t) {
    const bf16_t* hr = af + ((size_t)(b * 2 + t) * NND + n) * 64;
#pragma unroll 8
    for (int c = 0; c < 64; ++c) acc += (float)hr[c] * wr[t * 64 + c];
  }
  const float* er = emb + (size_t)n * 96;
#pragma unroll 8
  for (int e = 0; e < 96; ++e) acc += er[e] * wr[128 + e];
  out[((size_t)(b * 12 + p) * NND) + n] = acc;
}

// ---------------------------------------------------------------------------

static inline size_t alignup(size_t x) { return (x + 255) & ~(size_t)255; }

extern "C" void kernel_launch(void* const* d_in, const int* in_sizes, int n_in,
                              void* d_out, int out_size, void* d_ws, size_t ws_size,
                              hipStream_t stream) {
  (void)in_sizes; (void)n_in; (void)out_size; (void)ws_size;
  const float* x      = (const float*)d_in[0];
  const float* x_emb  = (const float*)d_in[1];
  const float* cheb   = (const float*)d_in[2];
  const float* theta1 = (const float*)d_in[3];
  const float* theta2 = (const float*)d_in[4];
  const float* w_t11  = (const float*)d_in[5];
  const float* b_t11  = (const float*)d_in[6];
  const float* w_t12  = (const float*)d_in[7];
  const float* b_t12  = (const float*)d_in[8];
  const float* w_t21  = (const float*)d_in[9];
  const float* b_t21  = (const float*)d_in[10];
  const float* w_t22  = (const float*)d_in[11];
  const float* b_t22  = (const float*)d_in[12];
  const float* w_tf   = (const float*)d_in[13];
  const float* b_tf   = (const float*)d_in[14];
  const float* fc_w   = (const float*)d_in[15];
  const float* fc_b   = (const float*)d_in[16];
  float* out = (float*)d_out;

  // workspace carve-up (~85 MB)
  char* ws = (char*)d_ws;
  size_t off = 0;
  bf16_t* chebB = (bf16_t*)(ws + off); off = alignup(off + (size_t)3 * NND * NND * 2);
  bf16_t* W12   = (bf16_t*)(ws + off); off = alignup(off + (size_t)128 * 64 * 2);
  bf16_t* W21   = (bf16_t*)(ws + off); off = alignup(off + (size_t)128 * 192 * 2);
  bf16_t* W22   = (bf16_t*)(ws + off); off = alignup(off + (size_t)128 * 64 * 2);
  bf16_t* WTF   = (bf16_t*)(ws + off); off = alignup(off + (size_t)128 * 192 * 2);
  bf16_t* A0    = (bf16_t*)(ws + off); off = alignup(off + (size_t)160 * 64 * NND * 2);
  bf16_t* A1    = (bf16_t*)(ws + off); off = alignup(off + (size_t)160 * 64 * NND * 2);
  float*  OACC  = (float*) (ws + off); off = alignup(off + (size_t)160 * NND * 16 * 4);
  bf16_t* SBUF  = (bf16_t*)(ws + off); off = alignup(off + (size_t)160 * NND * 16 * 2);
  bf16_t* MBUF  = (bf16_t*)(ws + off); off = alignup(off + (size_t)128 * NND * 64 * 2);
  bf16_t* AF    = (bf16_t*)(ws + off); off = alignup(off + (size_t)32  * NND * 64 * 2);

  // ---- preprocessing: bf16 casts
  k_cvt_bf16<<<dim3((3 * NND * NND + 255) / 256), dim3(256), 0, stream>>>(
      cheb, chebB, 3 * NND * NND);
  k_prep_w<<<dim3((128 * 64  + 255) / 256), dim3(256), 0, stream>>>(w_t12, W12, 16, 64);
  k_prep_w<<<dim3((128 * 192 + 255) / 256), dim3(256), 0, stream>>>(w_t21, W21, 64, 192);
  k_prep_w<<<dim3((128 * 64  + 255) / 256), dim3(256), 0, stream>>>(w_t22, W22, 16, 64);
  k_prep_w<<<dim3((128 * 192 + 255) / 256), dim3(256), 0, stream>>>(w_tf,  WTF, 64, 192);

  // ---- block 1 temporal conv (Cin=1): x -> A0 c-major, T 12->10
  k_tconv1<<<dim3(NND / 256, 160), dim3(256), 0, stream>>>(x, w_t11, b_t11, A0, 12, 10);

  // ---- spatial conv 1 (BT=160): ping-pong A0/A1, theta1 -> SBUF (s1)
  {
    bf16_t* ping = A0; bf16_t* pong = A1;
    for (int k = 0; k < 3; ++k) {
      k_spmm_wmma<<<dim3(4, 160), dim3(256), 0, stream>>>(
          ping, chebB + (size_t)k * NND * NND, pong);
      k_theta<<<dim3(64, 160), dim3(256), 0, stream>>>(
          pong, theta1 + (size_t)k * 64 * 16, OACC, SBUF, k);
      bf16_t* t = ping; ping = pong; pong = t;
    }
  }

  // ---- tconv w_t12: s1 (Ci=16) -> MBUF n-major, T 10->8
  k_tconv_wmma<<<dim3(NND / 32, 128), dim3(256), 0, stream>>>(
      SBUF, W12, b_t12, MBUF, (bf16_t*)nullptr, 10, 8, 16, 4, 64);

  // ---- tconv w_t21: MBUF (Ci=64) -> A0 c-major, T 8->6
  k_tconv_wmma<<<dim3(NND / 32, 96), dim3(256), 0, stream>>>(
      MBUF, W21, b_t21, (bf16_t*)nullptr, A0, 8, 6, 64, 6, 192);

  // ---- spatial conv 2 (BT=96): theta2 -> SBUF (s2)
  {
    bf16_t* ping = A0; bf16_t* pong = A1;
    for (int k = 0; k < 3; ++k) {
      k_spmm_wmma<<<dim3(4, 96), dim3(256), 0, stream>>>(
          ping, chebB + (size_t)k * NND * NND, pong);
      k_theta<<<dim3(64, 96), dim3(256), 0, stream>>>(
          pong, theta2 + (size_t)k * 64 * 16, OACC, SBUF, k);
      bf16_t* t = ping; ping = pong; pong = t;
    }
  }

  // ---- tconv w_t22: s2 (Ci=16) -> MBUF n-major, T 6->4
  k_tconv_wmma<<<dim3(NND / 32, 64), dim3(256), 0, stream>>>(
      SBUF, W22, b_t22, MBUF, (bf16_t*)nullptr, 6, 4, 16, 4, 64);

  // ---- tconv w_tf: MBUF (Ci=64) -> AF n-major, T 4->2
  k_tconv_wmma<<<dim3(NND / 32, 32), dim3(256), 0, stream>>>(
      MBUF, WTF, b_tf, AF, (bf16_t*)nullptr, 4, 2, 64, 6, 192);

  // ---- FC head -> out (B,12,1024,1) f32
  k_fc<<<dim3(16 * 12 * NND / 256), dim3(256), 0, stream>>>(
      AF, x_emb, fc_w, fc_b, out);
}
